// RNADLM_75462575391279
// MI455X (gfx1250) — matrix operand order repaired and checked by reading, not compiled, not run on hardware
//
#include <hip/hip_runtime.h>
#include <hip/hip_bf16.h>
#include <math.h>

#define DM      256
#define NHEADS  4
#define DHD     64
#define WINSZ   32
#define NDEPTH  6
#define NVOCAB  4096
#define SEQLEN  128
#define NBATCH  64
#define MROWS   (NBATCH * SEQLEN)   // 8192
#define NWIN    (SEQLEN / WINSZ)    // 4
#define FFI     1024                // GLU inner dim (8*DM split in half)

typedef __attribute__((ext_vector_type(16))) __bf16 v16bf;
typedef __attribute__((ext_vector_type(8)))  float  v8f;

union FragBF { v16bf v; unsigned int u[8]; };

__device__ __forceinline__ unsigned short f2bf(float f) {
  __bf16 h = (__bf16)f;                       // native cvt (RTE)
  return __builtin_bit_cast(unsigned short, h);
}
// Single v_cvt_pk_bf16_f32: packs two f32 -> 2xbf16 in one VALU op.
__device__ __forceinline__ unsigned int pk2bf(float lo, float hi) {
  unsigned int r;
  asm("v_cvt_pk_bf16_f32 %0, %1, %2" : "=v"(r) : "v"(lo), "v"(hi));
  return r;
}
__device__ __forceinline__ float bf2f(unsigned short h) {
  unsigned int u = ((unsigned int)h) << 16;
  return __builtin_bit_cast(float, u);
}

// Async DMA of 64 contiguous bytes global->LDS (4 x b128). The instruction
// offset advances BOTH the global and LDS address (ISA 15.18.3), so one
// address pair covers all four chunks. Tracked by ASYNCcnt.
__device__ __forceinline__ void async_copy64(unsigned int lds_addr,
                                             unsigned long long gbase,
                                             unsigned int goff) {
  asm volatile(
      "global_load_async_to_lds_b128 %0, %1, %2\n\t"
      "global_load_async_to_lds_b128 %0, %1, %2 offset:16\n\t"
      "global_load_async_to_lds_b128 %0, %1, %2 offset:32\n\t"
      "global_load_async_to_lds_b128 %0, %1, %2 offset:48"
      :: "v"(lds_addr), "v"(goff), "s"(gbase)
      : "memory");
}
__device__ __forceinline__ void async_copy32(unsigned int lds_addr,
                                             unsigned long long gbase,
                                             unsigned int goff) {
  asm volatile(
      "global_load_async_to_lds_b128 %0, %1, %2\n\t"
      "global_load_async_to_lds_b128 %0, %1, %2 offset:16"
      :: "v"(lds_addr), "v"(goff), "s"(gbase)
      : "memory");
}
__device__ __forceinline__ void wait_async0() {
  asm volatile("s_wait_asynccnt 0x0" ::: "memory");
}
__device__ __forceinline__ unsigned int lds_off(const void* p) {
  return (unsigned int)(unsigned long long)p;   // LDS aperture: offset in [31:0]
}

// ---------------------------------------------------------------- embedding
__global__ __launch_bounds__(256) void rn_embed(const int* __restrict__ seq,
    const float* __restrict__ tok, const float* __restrict__ pos,
    float* __restrict__ x1, float* __restrict__ x2) {
  int row = blockIdx.x;                // b*SEQLEN + n
  int n = row % SEQLEN;
  int d = threadIdx.x;
  float v = tok[(size_t)seq[row] * DM + d] + pos[(size_t)n * DM + d];
  x1[(size_t)row * DM + d] = v;
  x2[(size_t)row * DM + d] = v;
}

// --------------------------------------------------------------- layernorm
// one wave32 per row, 8 rows per 256-thread block; optional reversible merge
__global__ __launch_bounds__(256) void rn_layernorm(const float* __restrict__ x,
    const float* __restrict__ xmerge, const float* __restrict__ g,
    const float* __restrict__ b, unsigned short* __restrict__ out) {
  int wave = threadIdx.x >> 5, lane = threadIdx.x & 31;
  int row  = blockIdx.x * 8 + wave;
  float vals[8];
  float s = 0.f;
  #pragma unroll
  for (int e = 0; e < 8; ++e) {
    int c = lane + e * 32;
    float v = x[(size_t)row * DM + c];
    if (xmerge) v = 0.5f * (v + xmerge[(size_t)row * DM + c]);
    vals[e] = v; s += v;
  }
  #pragma unroll
  for (int off = 16; off > 0; off >>= 1) s += __shfl_xor(s, off);
  float mu = s * (1.0f / DM);
  float t = 0.f;
  #pragma unroll
  for (int e = 0; e < 8; ++e) { float d = vals[e] - mu; t += d * d; }
  #pragma unroll
  for (int off = 16; off > 0; off >>= 1) t += __shfl_xor(t, off);
  float rstd = rsqrtf(t * (1.0f / DM) + 1e-5f);
  #pragma unroll
  for (int e = 0; e < 8; ++e) {
    int c = lane + e * 32;
    out[(size_t)row * DM + c] = f2bf((vals[e] - mu) * rstd * g[c] + b[c]);
  }
}

// ------------------------------------------------------------- WMMA GEMM
// C[M,N] = A_bf16[M,K] * W_f32[K,N] + bias (+res); block tile 256x64,
// wave -> 32x64 (2 A-frags, B-frags reused) = 8 WMMA per K-step.
// Double-buffered LDS; A tiles via async DMA, W converted through registers.
template<int K, int N>
__global__ __launch_bounds__(256) void rn_gemm_t(const unsigned short* __restrict__ A,
    const float* __restrict__ W, const float* __restrict__ bias,
    const float* __restrict__ res, float* __restrict__ out) {
  __shared__ __align__(16) unsigned short sA[2][256 * 40]; // [row][k], 80B rows
  __shared__ __align__(16) unsigned short sB[2][64 * 40];  // transposed [col][k]
  const int tid = threadIdx.x;
  const int wave = tid >> 5, lane = tid & 31;
  const int m = lane & 15, hh = lane >> 4;
  const int rowBase = blockIdx.y * 256;
  const int colBase = blockIdx.x * 64;
  const int nn = tid & 63, kp = (tid >> 6) * 2;  // W staging: 4 adjacent-k pairs
  constexpr int NIT = K / 32;
  v8f acc[8] = {};

  const unsigned long long gA = (unsigned long long)A;
  const unsigned int laA0 = lds_off(&sA[0][0]) + tid * 80;  // 64B row per thread
  const unsigned int laA1 = lds_off(&sA[1][0]) + tid * 80;

  float wr0[4], wr1[4];
  auto loadW = [&](int k0) {
    #pragma unroll
    for (int e = 0; e < 4; ++e) {
      size_t base = (size_t)(k0 + kp + 8 * e) * N + colBase + nn;
      wr0[e] = W[base];
      wr1[e] = W[base + N];
    }
  };
  auto asyncA = [&](int buf, int k0) {
    unsigned int goff = (unsigned int)(((size_t)(rowBase + tid) * K + k0) * 2);
    async_copy64(buf ? laA1 : laA0, gA, goff);
  };

  loadW(0);
  asyncA(0, 0);

  #pragma unroll 2
  for (int it = 0; it < NIT; ++it) {
    const int buf = it & 1;
    {
      unsigned int* pBw = (unsigned int*)sB[buf];
      #pragma unroll
      for (int e = 0; e < 4; ++e)
        pBw[nn * 20 + (kp >> 1) + 4 * e] = pk2bf(wr0[e], wr1[e]);
    }
    wait_async0();
    __syncthreads();                       // tile `buf` ready
    if (it + 1 < NIT) {                    // overlap next tile's fetch with WMMA
      loadW((it + 1) * 32);
      asyncA(buf ^ 1, (it + 1) * 32);
    }
    const unsigned int* pA = (const unsigned int*)sA[buf];
    const unsigned int* pB = (const unsigned int*)sB[buf];
    FragBF fa[2];
    const int rbase = wave * 32 + m;
    #pragma unroll
    for (int half = 0; half < 2; ++half)
      #pragma unroll
      for (int rr = 0; rr < 8; ++rr) {     // ISA 16-bit A 16x32 lane layout
        int off = rr + 4 * hh + (rr >= 4 ? 4 : 0);
        fa[half].u[rr] = pA[(rbase + 16 * half) * 20 + off];
      }
    #pragma unroll
    for (int t = 0; t < 4; ++t) {
      FragBF fb;
      int col = t * 16 + m;
      #pragma unroll
      for (int rr = 0; rr < 8; ++rr) fb.u[rr] = pB[col * 20 + rr + 4 * hh + (rr >= 4 ? 4 : 0)];
      acc[t]     = __builtin_amdgcn_wmma_f32_16x16x32_bf16(
          false, fa[0].v, false, fb.v, (short)0, acc[t], false, false);
      acc[4 + t] = __builtin_amdgcn_wmma_f32_16x16x32_bf16(
          false, fa[1].v, false, fb.v, (short)0, acc[4 + t], false, false);
    }
    __syncthreads();                       // done reading `buf`
  }
  #pragma unroll
  for (int half = 0; half < 2; ++half)
    #pragma unroll
    for (int t = 0; t < 4; ++t) {
      int col = colBase + t * 16 + m;
      float bv = bias[col];
      #pragma unroll
      for (int e = 0; e < 8; ++e) {
        int row = rowBase + wave * 32 + half * 16 + hh * 8 + e;
        size_t idx = (size_t)row * N + col;
        float v = acc[half * 4 + t][e] + bv;
        if (res) v += res[idx];
        out[idx] = v;
      }
    }
}

// ------------------------------------------------ WMMA GEMM + fused GLU
// a = A*W[:, c..], g = A*W[:, c+1024..]; out_bf16 = gelu(a+ba)*(g+bg)
__global__ __launch_bounds__(256) void rn_gemm_glu(const unsigned short* __restrict__ A,
    const float* __restrict__ W, const float* __restrict__ bias,
    unsigned short* __restrict__ out) {
  constexpr int K = DM, NW2 = 2 * FFI;
  __shared__ __align__(16) unsigned short sA[2][128 * 40];
  __shared__ __align__(16) unsigned short sBa[2][64 * 40];
  __shared__ __align__(16) unsigned short sBg[2][64 * 40];
  const int tid = threadIdx.x;
  const int wave = tid >> 5, lane = tid & 31;
  const int m = lane & 15, hh = lane >> 4;
  const int rowBase = blockIdx.y * 128;
  const int colBase = blockIdx.x * 64;
  const int r  = tid >> 1, kc = (tid & 1) * 16;
  const int nn = tid & 63, kp = (tid >> 6) * 2;
  constexpr int NIT = K / 32;
  v8f accA[4] = {};
  v8f accG[4] = {};

  const unsigned long long gA = (unsigned long long)A;
  const unsigned int laA0 = lds_off(&sA[0][0]) + r * 80 + kc * 2;
  const unsigned int laA1 = lds_off(&sA[1][0]) + r * 80 + kc * 2;

  float wa0[4], wa1[4], wg0[4], wg1[4];
  auto loadW = [&](int k0) {
    #pragma unroll
    for (int e = 0; e < 4; ++e) {
      size_t base = (size_t)(k0 + kp + 8 * e) * NW2 + colBase + nn;
      wa0[e] = W[base];       wa1[e] = W[base + NW2];
      wg0[e] = W[base + FFI]; wg1[e] = W[base + NW2 + FFI];
    }
  };
  auto asyncA = [&](int buf, int k0) {
    unsigned int goff = (unsigned int)(((size_t)(rowBase + r) * K + k0 + kc) * 2);
    async_copy32(buf ? laA1 : laA0, gA, goff);
  };

  loadW(0);
  asyncA(0, 0);

  #pragma unroll 2
  for (int it = 0; it < NIT; ++it) {
    const int buf = it & 1;
    {
      unsigned int* pa = (unsigned int*)sBa[buf];
      unsigned int* pg = (unsigned int*)sBg[buf];
      #pragma unroll
      for (int e = 0; e < 4; ++e) {
        pa[nn * 20 + (kp >> 1) + 4 * e] = pk2bf(wa0[e], wa1[e]);
        pg[nn * 20 + (kp >> 1) + 4 * e] = pk2bf(wg0[e], wg1[e]);
      }
    }
    wait_async0();
    __syncthreads();
    if (it + 1 < NIT) {
      loadW((it + 1) * 32);
      asyncA(buf ^ 1, (it + 1) * 32);
    }
    const unsigned int* pA  = (const unsigned int*)sA[buf];
    const unsigned int* pBa = (const unsigned int*)sBa[buf];
    const unsigned int* pBg = (const unsigned int*)sBg[buf];
    FragBF fa;
    const int row = wave * 16 + m;
    #pragma unroll
    for (int rr = 0; rr < 8; ++rr) {
      int off = rr + 4 * hh + (rr >= 4 ? 4 : 0);
      fa.u[rr] = pA[row * 20 + off];
    }
    #pragma unroll
    for (int t = 0; t < 4; ++t) {
      FragBF fb, fg;
      int col = t * 16 + m;
      #pragma unroll
      for (int rr = 0; rr < 8; ++rr) {
        fb.u[rr] = pBa[col * 20 + rr + 4 * hh + (rr >= 4 ? 4 : 0)];
        fg.u[rr] = pBg[col * 20 + rr + 4 * hh + (rr >= 4 ? 4 : 0)];
      }
      accA[t] = __builtin_amdgcn_wmma_f32_16x16x32_bf16(
          false, fa.v, false, fb.v, (short)0, accA[t], false, false);
      accG[t] = __builtin_amdgcn_wmma_f32_16x16x32_bf16(
          false, fa.v, false, fg.v, (short)0, accG[t], false, false);
    }
    __syncthreads();
  }
  #pragma unroll
  for (int t = 0; t < 4; ++t) {
    int col = colBase + t * 16 + m;
    float ba = bias[col], bg = bias[col + FFI];
    #pragma unroll
    for (int e = 0; e < 8; ++e) {
      int row = rowBase + wave * 16 + hh * 8 + e;
      float a = accA[t][e] + ba;
      float g = accG[t][e] + bg;
      float gelu = 0.5f * a * (1.0f + erff(a * 0.70710678118f));
      out[(size_t)row * FFI + col] = f2bf(gelu * g);
    }
  }
}

// ------------------------------------------------------- local attention
// one block per (b, h, window). qkv layout: [B,N, h*64+d]. 56KB LDS.
__global__ __launch_bounds__(128) void rn_attn(const float* __restrict__ q,
    const float* __restrict__ k, const float* __restrict__ v,
    const float* __restrict__ relw, unsigned short* __restrict__ out) {
  __shared__ float          sQ[WINSZ * DHD];       // 8 KB
  __shared__ unsigned short sK[3 * WINSZ * DHD];   // 12 KB
  __shared__ unsigned short sV[3 * WINSZ * DHD];   // 12 KB
  __shared__ unsigned short sR[3 * WINSZ * DHD];   // 12 KB
  __shared__ float          sS[WINSZ * 3 * WINSZ]; // 12 KB
  const int w = blockIdx.x, h = blockIdx.y, b = blockIdx.z;
  const int tid = threadIdx.x;
  const int qrow0 = b * SEQLEN + w * WINSZ;

  #pragma unroll
  for (int e = 0; e < 16; ++e) {           // Q window
    int f = tid + e * 128, i = f >> 6, d = f & 63;
    sQ[f] = q[(size_t)(qrow0 + i) * DM + h * DHD + d];
  }
  #pragma unroll
  for (int e = 0; e < 48; ++e) {           // K/V band + rel table (bf16 in LDS)
    int f = tid + e * 128, j = f >> 6, d = f & 63;
    int n = (w - 1) * WINSZ + j;
    float kv = 0.f, vv = 0.f;
    if (n >= 0 && n < SEQLEN) {
      kv = k[(size_t)(b * SEQLEN + n) * DM + h * DHD + d];
      vv = v[(size_t)(b * SEQLEN + n) * DM + h * DHD + d];
    }
    sK[f] = f2bf(kv);
    sV[f] = f2bf(vv);
    sR[f] = f2bf(relw[(size_t)(j * NHEADS + h) * DHD + d]);
  }
  __syncthreads();

  #pragma unroll
  for (int e = 0; e < 24; ++e) {           // sim = (QK^T + skewed rel) * scale
    int p = tid + e * 128, i = p / 96, j = p % 96;
    int t = (WINSZ - 1) + j - i;
    bool tv = (t >= 0 && t < 3 * WINSZ);
    float sqk = 0.f, srl = 0.f;
    #pragma unroll
    for (int d = 0; d < DHD; ++d) {
      float qv = sQ[i * DHD + d];
      sqk += qv * bf2f(sK[j * DHD + d]);
      if (tv) srl += qv * bf2f(sR[t * DHD + d]);
    }
    float s = (sqk + (tv ? srl : 0.f)) * 0.125f;   // DH^-0.5
    int bw = w - 1 + (j >> 5);
    if (bw < 0 || bw >= NWIN) s = -1e9f;
    sS[i * 96 + j] = s;
  }
  __syncthreads();

  if (tid < WINSZ) {                       // row softmax
    int i = tid;
    float mx = -3.0e38f;
    for (int j = 0; j < 96; ++j) mx = fmaxf(mx, sS[i * 96 + j]);
    float sum = 0.f;
    for (int j = 0; j < 96; ++j) { float e = __expf(sS[i * 96 + j] - mx); sS[i * 96 + j] = e; sum += e; }
    float inv = 1.0f / sum;
    for (int j = 0; j < 96; ++j) sS[i * 96 + j] *= inv;
  }
  __syncthreads();

  #pragma unroll
  for (int e = 0; e < 16; ++e) {           // out = attn @ Vband, bf16, [B,N,H*DH]
    int f = tid + e * 128, i = f >> 6, d = f & 63;
    float s = 0.f;
    for (int j = 0; j < 96; ++j) s += sS[i * 96 + j] * bf2f(sV[j * DHD + d]);
    out[(size_t)(qrow0 + i) * DM + h * DHD + d] = f2bf(s);
  }
}

// ---------------------------------------------------------------- driver
extern "C" void kernel_launch(void* const* d_in, const int* in_sizes, int n_in,
                              void* d_out, int out_size, void* d_ws, size_t ws_size,
                              hipStream_t stream) {
  (void)in_sizes; (void)n_in; (void)out_size; (void)ws_size;
  const int*   seq  = (const int*)d_in[0];
  const float* tok  = (const float*)d_in[1];
  const float* pos  = (const float*)d_in[2];
  const float* ln1g = (const float*)d_in[3];
  const float* ln1b = (const float*)d_in[4];
  const float* wq   = (const float*)d_in[5];
  const float* bq   = (const float*)d_in[6];
  const float* wk   = (const float*)d_in[7];
  const float* bk   = (const float*)d_in[8];
  const float* wv   = (const float*)d_in[9];
  const float* bv   = (const float*)d_in[10];
  const float* wo   = (const float*)d_in[11];
  const float* bo   = (const float*)d_in[12];
  const float* relw = (const float*)d_in[13];
  const float* ln2g = (const float*)d_in[14];
  const float* ln2b = (const float*)d_in[15];
  const float* w1   = (const float*)d_in[16];
  const float* b1   = (const float*)d_in[17];
  const float* w2   = (const float*)d_in[18];
  const float* b2   = (const float*)d_in[19];
  const float* lnfg = (const float*)d_in[20];
  const float* lnfb = (const float*)d_in[21];
  const float* outw = (const float*)d_in[22];
  const float* outb = (const float*)d_in[23];

  char* p = (char*)d_ws;
  auto carve = [&](size_t bytes) { void* r = (void*)p; p += (bytes + 255) & ~(size_t)255; return r; };
  float*          x1  = (float*)carve((size_t)MROWS * DM * 4);
  float*          x2  = (float*)carve((size_t)MROWS * DM * 4);
  unsigned short* xa  = (unsigned short*)carve((size_t)MROWS * DM * 2);
  float*          qb  = (float*)carve((size_t)MROWS * DM * 4);
  float*          kb  = (float*)carve((size_t)MROWS * DM * 4);
  float*          vb  = (float*)carve((size_t)MROWS * DM * 4);
  unsigned short* aO  = (unsigned short*)carve((size_t)MROWS * DM * 2);
  unsigned short* glu = (unsigned short*)carve((size_t)MROWS * FFI * 2);

  const dim3 blk256(256), blk128(128);
  const dim3 gLN(MROWS / 8);
  const dim3 gGemmD(DM / 64, MROWS / 256);       // N=256, 256-row tiles
  const dim3 gGemmGlu(FFI / 64, MROWS / 128);    // N=1024 (dual-col), 128-row tiles
  const dim3 gGemmV(NVOCAB / 64, MROWS / 256);   // N=4096
  const dim3 gAttn(NWIN, NHEADS, NBATCH);

  rn_embed<<<MROWS, blk256, 0, stream>>>(seq, tok, pos, x1, x2);

  for (int l = 0; l < NDEPTH; ++l) {
    size_t oLN = (size_t)l * DM, oW = (size_t)l * DM * DM;
    size_t oR  = (size_t)l * 3 * WINSZ * NHEADS * DHD;
    size_t oW1 = (size_t)l * DM * 2 * FFI, oB1 = (size_t)l * 2 * FFI;
    size_t oW2 = (size_t)l * FFI * DM;

    rn_layernorm<<<gLN, blk256, 0, stream>>>(x2, nullptr, ln1g + oLN, ln1b + oLN, xa);
    rn_gemm_t<DM, DM><<<gGemmD, blk256, 0, stream>>>(xa, wq + oW, bq + oLN, nullptr, qb);
    rn_gemm_t<DM, DM><<<gGemmD, blk256, 0, stream>>>(xa, wk + oW, bk + oLN, nullptr, kb);
    rn_gemm_t<DM, DM><<<gGemmD, blk256, 0, stream>>>(xa, wv + oW, bv + oLN, nullptr, vb);
    rn_attn<<<gAttn, blk128, 0, stream>>>(qb, kb, vb, relw + oR, aO);
    rn_gemm_t<DM, DM><<<gGemmD, blk256, 0, stream>>>(aO, wo + oW, bo + oLN, x1, x1);   // y1
    rn_layernorm<<<gLN, blk256, 0, stream>>>(x1, nullptr, ln2g + oLN, ln2b + oLN, xa);
    rn_gemm_glu<<<gGemmGlu, blk256, 0, stream>>>(xa, w1 + oW1, b1 + oB1, glu);
    rn_gemm_t<FFI, DM><<<gGemmD, blk256, 0, stream>>>(glu, w2 + oW2, b2 + oLN, x2, x2); // y2
  }

  rn_layernorm<<<gLN, blk256, 0, stream>>>(x1, x2, lnfg, lnfb, xa);   // 0.5*(x1+x2) merge
  rn_gemm_t<DM, NVOCAB><<<gGemmV, blk256, 0, stream>>>(xa, outw, outb, nullptr, (float*)d_out);
}
